// AxialAttention3d_87213605912631
// MI455X (gfx1250) — matrix operand order repaired and checked by reading, not compiled, not run on hardware
//
#include <hip/hip_runtime.h>

// ---------------- constants ----------------
#define NTOT 2048      // B*H*W
#define LL   32        // L
#define CIN  64
#define OC   128       // 2*OUT
#define NG   8         // GROUPS
#define BLK  256

typedef __attribute__((ext_vector_type(16))) __bf16 v16bf;
typedef __attribute__((ext_vector_type(8)))  float  v8f;

// workspace layout (float offsets)
#define WS_QKV   0            // N*128*32 = 8388608  (later overwritten with pre-BN out)
#define WS_P1    8388608      // 2048*256 qkv stats partials
#define WS_PSIM  (WS_P1 + 524288)     // 2048*48
#define WS_POUT  (WS_PSIM + 98304)    // 2048*256
#define WS_BNQ   (WS_POUT + 524288)   // 256: scale[128], shift[128]
#define WS_BNS   (WS_BNQ + 256)       // 48 : scale[24],  shift[24]
#define WS_BNO   (WS_BNS + 48)        // 256

// ---------------- WMMA helpers ----------------
// A 16x32 bf16 fragment, LDS row-major [m][k], row stride ld (bf16 elems).
// ISA layout: lanes0-15 m=0..15 hold K 0..7 (v0..3) & 16..23 (v4..7); lanes16-31 hold K 8..15 & 24..31.
static __device__ inline v16bf load_frag_a(const __bf16* base, int ld) {
  int lane = threadIdx.x & 31;
  int lo = lane & 15, hi = lane >> 4;
  union { v16bf v; unsigned u[8]; } f;
  const __bf16* row = base + lo * ld;
#pragma unroll
  for (int r = 0; r < 8; ++r) {
    int kp = (r < 4) ? (2 * r + hi * 8) : (2 * (r - 4) + 16 + hi * 8);
    f.u[r] = *reinterpret_cast<const unsigned*>(row + kp);
  }
  return f.v;
}
// B 32x16 bf16 fragment from transposed LDS tile Bt[n][k], stride ld.
// lanes0-15: n=lane, K 0..7 (v0..3) & 16..23 (v4..7); lanes16-31: K 8..15 & 24..31.
static __device__ inline v16bf load_frag_b(const __bf16* baseT, int ld) {
  int lane = threadIdx.x & 31;
  int lo = lane & 15, hi = lane >> 4;
  union { v16bf v; unsigned u[8]; } f;
  const __bf16* row = baseT + lo * ld;
#pragma unroll
  for (int r = 0; r < 8; ++r) {
    int kp = (r < 4) ? (2 * r + hi * 8) : (2 * (r - 4) + 16 + hi * 8);
    f.u[r] = *reinterpret_cast<const unsigned*>(row + kp);
  }
  return f.v;
}
static __device__ inline v8f wmma_bf16(v16bf a, v16bf b, v8f c) {
  return __builtin_amdgcn_wmma_f32_16x16x32_bf16(false, a, false, b, (short)0, c, false, false);
}

// ---------------- K1: qkv = W(128x64) @ X(64x32) per n, + stats ----------------
__global__ void k1_qkv(const float* __restrict__ x, const float* __restrict__ w,
                       float* __restrict__ ws) {
  __shared__ __bf16 Wbf[OC * CIN];   // [o][c]
  __shared__ __bf16 XT[LL * CIN];    // [l][c]  (B transposed)
  __shared__ float  Qk[OC * LL];
  int t = threadIdx.x, n = blockIdx.x;
  int b = n >> 10, hw = n & 1023;
  const float* xb = x + (size_t)b * 64 * 32768 + (size_t)hw * 32;
  for (int i = t; i < OC * CIN; i += BLK) Wbf[i] = (__bf16)w[i];
  for (int i = t; i < CIN * LL; i += BLK) {
    int c = i >> 5, l = i & 31;
    XT[l * CIN + c] = (__bf16)xb[(size_t)c * 32768 + l];
  }
  __syncthreads();
  int wv = t >> 5, lane = t & 31, lo = lane & 15, hi = lane >> 4;
  v8f z = {0.f,0.f,0.f,0.f,0.f,0.f,0.f,0.f};
  v16bf a0 = load_frag_a(Wbf + wv * 16 * CIN, CIN);
  v16bf a1 = load_frag_a(Wbf + wv * 16 * CIN + 32, CIN);
#pragma unroll
  for (int nt = 0; nt < 2; ++nt) {
    v16bf b0 = load_frag_b(XT + nt * 16 * CIN, CIN);
    v16bf b1 = load_frag_b(XT + nt * 16 * CIN + 32, CIN);
    v8f acc = wmma_bf16(a0, b0, z);
    acc = wmma_bf16(a1, b1, acc);
#pragma unroll
    for (int r = 0; r < 8; ++r)
      Qk[(wv * 16 + r + hi * 8) * LL + nt * 16 + lo] = acc[r];
  }
  __syncthreads();
  float* q = ws + WS_QKV + (size_t)n * (OC * LL);
  for (int u = 0; u < 16; ++u) q[t * 16 + u] = Qk[t * 16 + u];
  if (t < OC) {
    float s = 0.f, s2 = 0.f;
    for (int l = 0; l < LL; ++l) { float v = Qk[t * LL + l]; s += v; s2 += v * v; }
    ws[WS_P1 + (size_t)n * 256 + t] = s;
    ws[WS_P1 + (size_t)n * 256 + 128 + t] = s2;
  }
}

// ---------------- deterministic BN finalize: partial[blk][2*nch] -> scale/shift ----------------
__global__ void bn_finalize(const float* __restrict__ partial, int nblk, int nch,
                            float inv_count, const float* __restrict__ gamma,
                            const float* __restrict__ beta, float* __restrict__ params) {
  __shared__ float red[512];
  int t = threadIdx.x, slots = 2 * nch;
  if (t < slots) {
    float s = 0.f;
    for (int k = 0; k < nblk; ++k) s += partial[(size_t)k * slots + t];
    red[t] = s;
  }
  __syncthreads();
  if (t < nch) {
    float mean = red[t] * inv_count;
    float var  = red[nch + t] * inv_count - mean * mean;
    float sc = gamma[t] * rsqrtf(var + 1e-5f);
    params[t] = sc;
    params[nch + t] = beta[t] - mean * sc;
  }
}

// ---------------- K3: qe/ke (VALU) + qk (WMMA) -> sim stats ----------------
__global__ void k3_simstats(const float* __restrict__ rel, float* __restrict__ ws) {
  __shared__ float  relS[16 * 63];
  __shared__ float  bnq[256];
  __shared__ float  QKV[OC * LL];
  __shared__ __bf16 QE[NG][LL][LL];  // [g][b][a] : qe[g,a,b]
  __shared__ __bf16 KE[NG][LL][LL];
  __shared__ float  QK[NG][LL][LL];  // [g][i][j]
  __shared__ float  pA[256], pB[256];
  int t = threadIdx.x, n = blockIdx.x;
  for (int i = t; i < 16 * 63; i += BLK) relS[i] = rel[i];
  for (int i = t; i < 256; i += BLK) bnq[i] = ws[WS_BNQ + i];
  __syncthreads();
  for (int i = t; i < OC * LL; i += BLK) {
    int o = i >> 5;
    QKV[i] = ws[WS_QKV + (size_t)n * (OC * LL) + i] * bnq[o] + bnq[128 + o];
  }
  __syncthreads();
  int g = t >> 5, bcol = t & 31;
  float sq = 0.f, sq2 = 0.f, sk = 0.f, sk2 = 0.f;
  for (int a = 0; a < LL; ++a) {
    int d = a - bcol + 31;
    float vq = 0.f, vk = 0.f;
#pragma unroll
    for (int tt = 0; tt < 4; ++tt) {
      vq += QKV[(16 * g + tt) * LL + a]     * relS[tt * 63 + d];
      vk += QKV[(16 * g + 4 + tt) * LL + a] * relS[(4 + tt) * 63 + d];
    }
    QE[g][bcol][a] = (__bf16)vq; KE[g][bcol][a] = (__bf16)vk;
    sq += vq; sq2 += vq * vq; sk += vk; sk2 += vk * vk;
  }
  pA[t] = sq; pB[t] = sq2; __syncthreads();
  if (t < NG) {
    float s = 0.f, s2 = 0.f;
    for (int u = 0; u < 32; ++u) { s += pA[t * 32 + u]; s2 += pB[t * 32 + u]; }
    ws[WS_PSIM + (size_t)n * 48 + 8 + t] = s;
    ws[WS_PSIM + (size_t)n * 48 + 24 + 8 + t] = s2;
  }
  __syncthreads();
  pA[t] = sk; pB[t] = sk2; __syncthreads();
  if (t < NG) {
    float s = 0.f, s2 = 0.f;
    for (int u = 0; u < 32; ++u) { s += pA[t * 32 + u]; s2 += pB[t * 32 + u]; }
    ws[WS_PSIM + (size_t)n * 48 + 16 + t] = s;
    ws[WS_PSIM + (size_t)n * 48 + 24 + 16 + t] = s2;
  }
  __syncthreads();
  // qk = qe^T * ke per group, wave g does its 4 tiles
  int wv = t >> 5, lane = t & 31, lo = lane & 15, hi = lane >> 4;
  v8f z = {0.f,0.f,0.f,0.f,0.f,0.f,0.f,0.f};
  v16bf a0 = load_frag_a(&QE[wv][0][0], LL);
  v16bf a1 = load_frag_a(&QE[wv][16][0], LL);
  v16bf b0 = load_frag_b(&KE[wv][0][0], LL);
  v16bf b1 = load_frag_b(&KE[wv][16][0], LL);
  v8f c00 = wmma_bf16(a0, b0, z), c01 = wmma_bf16(a0, b1, z);
  v8f c10 = wmma_bf16(a1, b0, z), c11 = wmma_bf16(a1, b1, z);
#pragma unroll
  for (int r = 0; r < 8; ++r) {
    QK[wv][r + hi * 8][lo]           = c00[r];
    QK[wv][r + hi * 8][16 + lo]      = c01[r];
    QK[wv][16 + r + hi * 8][lo]      = c10[r];
    QK[wv][16 + r + hi * 8][16 + lo] = c11[r];
  }
  __syncthreads();
  float s = 0.f, s2 = 0.f;
  for (int j = 0; j < LL; ++j) { float v = QK[g][bcol][j]; s += v; s2 += v * v; }
  pA[t] = s; pB[t] = s2; __syncthreads();
  if (t < NG) {
    float a = 0.f, a2 = 0.f;
    for (int u = 0; u < 32; ++u) { a += pA[t * 32 + u]; a2 += pB[t * 32 + u]; }
    ws[WS_PSIM + (size_t)n * 48 + t] = a;
    ws[WS_PSIM + (size_t)n * 48 + 24 + t] = a2;
  }
}

// ---------------- K5: full attention, writes pre-BN out over WS_QKV + out stats ----------------
__global__ void k5_attn(const float* __restrict__ rel, float* __restrict__ ws) {
  __shared__ float  relS[16 * 63];
  __shared__ float  bnq[256];
  __shared__ float  bns[48];
  __shared__ float  QKV[OC * LL];
  __shared__ __bf16 QE[NG][LL][LL];
  __shared__ __bf16 KE[NG][LL][LL];
  __shared__ float  QK[NG][LL][LL];
  __shared__ __bf16 SIM[NG][LL][LL];     // [g][i][j]
  __shared__ __bf16 VB[NG][16][LL];      // B^T tile for am: [g][c][j], rows 8..15 zero
  __shared__ float  AM[NG][8][LL];       // [g][c][i]
  __shared__ float  pA[256];
  int t = threadIdx.x, n = blockIdx.x;
  for (int i = t; i < 16 * 63; i += BLK) relS[i] = rel[i];
  for (int i = t; i < 256; i += BLK) bnq[i] = ws[WS_BNQ + i];
  if (t < 48) bns[t] = ws[WS_BNS + t];
  __syncthreads();
  for (int i = t; i < OC * LL; i += BLK) {
    int o = i >> 5;
    QKV[i] = ws[WS_QKV + (size_t)n * (OC * LL) + i] * bnq[o] + bnq[128 + o];
  }
  __syncthreads();
  // v tile as bf16, padded to 16 rows
  for (int i = t; i < NG * 16 * LL; i += BLK) {
    int g2 = i >> 9, c2 = (i >> 5) & 15, j2 = i & 31;
    float v = (c2 < 8) ? QKV[(16 * g2 + 8 + c2) * LL + j2] : 0.f;
    VB[g2][c2][j2] = (__bf16)v;
  }
  // qe / ke
  int g = t >> 5, bcol = t & 31;
  for (int a = 0; a < LL; ++a) {
    int d = a - bcol + 31;
    float vq = 0.f, vk = 0.f;
#pragma unroll
    for (int tt = 0; tt < 4; ++tt) {
      vq += QKV[(16 * g + tt) * LL + a]     * relS[tt * 63 + d];
      vk += QKV[(16 * g + 4 + tt) * LL + a] * relS[(4 + tt) * 63 + d];
    }
    QE[g][bcol][a] = (__bf16)vq; KE[g][bcol][a] = (__bf16)vk;
  }
  __syncthreads();
  // qk WMMA
  int wv = t >> 5, lane = t & 31, lo = lane & 15, hi = lane >> 4;
  v8f z = {0.f,0.f,0.f,0.f,0.f,0.f,0.f,0.f};
  {
    v16bf a0 = load_frag_a(&QE[wv][0][0], LL);
    v16bf a1 = load_frag_a(&QE[wv][16][0], LL);
    v16bf b0 = load_frag_b(&KE[wv][0][0], LL);
    v16bf b1 = load_frag_b(&KE[wv][16][0], LL);
    v8f c00 = wmma_bf16(a0, b0, z), c01 = wmma_bf16(a0, b1, z);
    v8f c10 = wmma_bf16(a1, b0, z), c11 = wmma_bf16(a1, b1, z);
#pragma unroll
    for (int r = 0; r < 8; ++r) {
      QK[wv][r + hi * 8][lo]           = c00[r];
      QK[wv][r + hi * 8][16 + lo]      = c01[r];
      QK[wv][16 + r + hi * 8][lo]      = c10[r];
      QK[wv][16 + r + hi * 8][16 + lo] = c11[r];
    }
  }
  __syncthreads();
  // sim = bn(qk)+bn(qe)+bn(ke), softmax over j   (thread owns row (g, i=bcol))
  {
    float row[LL]; float mx = -1e30f;
    float sA = bns[g], hA = bns[24 + g];
    float sB = bns[8 + g], hB = bns[32 + g];
    float sC = bns[16 + g], hC = bns[40 + g];
    for (int j = 0; j < LL; ++j) {
      float v = QK[g][bcol][j] * sA + hA
              + (float)QE[g][j][bcol] * sB + hB
              + (float)KE[g][j][bcol] * sC + hC;
      row[j] = v; mx = fmaxf(mx, v);
    }
    float sum = 0.f;
    for (int j = 0; j < LL; ++j) { float e = __expf(row[j] - mx); row[j] = e; sum += e; }
    float inv = 1.f / sum;
    for (int j = 0; j < LL; ++j) SIM[g][bcol][j] = (__bf16)(row[j] * inv);
  }
  __syncthreads();
  // am = sim * v^T  (wave g: M=32 -> 2 tiles, N=16 (8 valid), K=32)
  {
    v16bf sa0 = load_frag_a(&SIM[wv][0][0], LL);
    v16bf sa1 = load_frag_a(&SIM[wv][16][0], LL);
    v16bf vb  = load_frag_b(&VB[wv][0][0], LL);
    v8f m0 = wmma_bf16(sa0, vb, z), m1 = wmma_bf16(sa1, vb, z);
    if (lo < 8) {
#pragma unroll
      for (int r = 0; r < 8; ++r) {
        AM[wv][lo][r + hi * 8]      = m0[r];
        AM[wv][lo][16 + r + hi * 8] = m1[r];
      }
    }
  }
  __syncthreads();
  // ame + assemble out channels (overwrite qkv region) + stats
  int p = t >> 2, sub = t & 3;      // p = g*8+c
  int g3 = p >> 3, c3 = p & 7;
  float s_am = 0.f, s_am2 = 0.f, s_ae = 0.f, s_ae2 = 0.f;
  float* outp = ws + WS_QKV + (size_t)n * (OC * LL);
  for (int i = sub * 8; i < sub * 8 + 8; ++i) {
    float ae = 0.f;
    const float* rb = &relS[(8 + c3) * 63 + i + 31];
    for (int j = 0; j < LL; ++j) ae += (float)SIM[g3][i][j] * rb[-j];
    float am = AM[g3][c3][i];
    outp[(2 * p) * LL + i] = am;
    outp[(2 * p + 1) * LL + i] = ae;
    s_am += am; s_am2 += am * am; s_ae += ae; s_ae2 += ae * ae;
  }
  float* po = ws + WS_POUT + (size_t)n * 256;
  pA[t] = s_am; __syncthreads();
  if (t < 64) po[2 * t] = pA[4 * t] + pA[4 * t + 1] + pA[4 * t + 2] + pA[4 * t + 3];
  __syncthreads(); pA[t] = s_am2; __syncthreads();
  if (t < 64) po[128 + 2 * t] = pA[4 * t] + pA[4 * t + 1] + pA[4 * t + 2] + pA[4 * t + 3];
  __syncthreads(); pA[t] = s_ae; __syncthreads();
  if (t < 64) po[2 * t + 1] = pA[4 * t] + pA[4 * t + 1] + pA[4 * t + 2] + pA[4 * t + 3];
  __syncthreads(); pA[t] = s_ae2; __syncthreads();
  if (t < 64) po[128 + 2 * t + 1] = pA[4 * t] + pA[4 * t + 1] + pA[4 * t + 2] + pA[4 * t + 3];
}

// ---------------- K7: out BN, pair-sum, scatter to (B,64,H,W,L) ----------------
__global__ void k7_out(const float* __restrict__ ws, float* __restrict__ out) {
  int t = threadIdx.x, n = blockIdx.x;
  int b = n >> 10, hw = n & 1023;
  const float* p  = ws + WS_QKV + (size_t)n * (OC * LL);
  const float* sc = ws + WS_BNO;
  const float* sh = ws + WS_BNO + 128;
  int base = t * 8;
  int o = base >> 5, l0 = base & 31;
  float s0 = sc[2 * o], h0 = sh[2 * o], s1 = sc[2 * o + 1], h1 = sh[2 * o + 1];
  float* ob = out + (size_t)(b * 64 + o) * 32768 + (size_t)hw * 32;
  for (int u = 0; u < 8; ++u) {
    int l = l0 + u;
    ob[l] = p[(2 * o) * LL + l] * s0 + h0 + p[(2 * o + 1) * LL + l] * s1 + h1;
  }
}

// ---------------- launcher ----------------
extern "C" void kernel_launch(void* const* d_in, const int* in_sizes, int n_in,
                              void* d_out, int out_size, void* d_ws, size_t ws_size,
                              hipStream_t stream) {
  (void)in_sizes; (void)n_in; (void)out_size; (void)ws_size;
  const float* x     = (const float*)d_in[0];
  const float* w_qkv = (const float*)d_in[1];
  const float* gq    = (const float*)d_in[2];
  const float* bq    = (const float*)d_in[3];
  const float* gs    = (const float*)d_in[4];
  const float* bs    = (const float*)d_in[5];
  const float* go    = (const float*)d_in[6];
  const float* bo    = (const float*)d_in[7];
  const float* rel   = (const float*)d_in[8];
  float* ws  = (float*)d_ws;
  float* out = (float*)d_out;

  k1_qkv<<<NTOT, BLK, 0, stream>>>(x, w_qkv, ws);
  bn_finalize<<<1, BLK, 0, stream>>>(ws + WS_P1, NTOT, 128, 1.f / 65536.f, gq, bq, ws + WS_BNQ);
  k3_simstats<<<NTOT, BLK, 0, stream>>>(rel, ws);
  bn_finalize<<<1, BLK, 0, stream>>>(ws + WS_PSIM, NTOT, 24, 1.f / 2097152.f, gs, bs, ws + WS_BNS);
  k5_attn<<<NTOT, BLK, 0, stream>>>(rel, ws);
  bn_finalize<<<1, BLK, 0, stream>>>(ws + WS_POUT, NTOT, 128, 1.f / 65536.f, go, bo, ws + WS_BNO);
  k7_out<<<NTOT, BLK, 0, stream>>>(ws, out);
}
